// SelfAttentionBlock_7138235646557
// MI455X (gfx1250) — compile-verified, matrix operand order checked by reading, not run on hardware
//
#include <hip/hip_runtime.h>
#include <hip/hip_bf16.h>

typedef __attribute__((ext_vector_type(16))) __bf16 v16bf;
typedef __attribute__((ext_vector_type(8)))  float  v8f;
typedef __attribute__((ext_vector_type(2)))  float  v2f;

namespace {

constexpr int kC  = 64;          // channels
constexpr int kD  = 32;          // depth
constexpr int kHW = 4096;        // H*W
constexpr int kTW = 8;           // hw columns per workgroup
constexpr int kQS = 264;         // q/k/v per-(head,hw) plane stride (f32); 8-dword aligned, bank skew 8
constexpr int kXR = 80;          // swizzled bf16 row stride; rows 32B-aligned, bank skew 40 dwords
constexpr int kPS = 34;          // P scratch row stride (f32); keeps b64 pair loads 8B-aligned

struct __attribute__((aligned(4))) bf2 { __bf16 x, y; };

__device__ __forceinline__ void store_bf2(__bf16* p, float a, float b) {
  bf2 t; t.x = (__bf16)a; t.y = (__bf16)b;
  *(bf2*)p = t;
}

// Position of source index k inside a swizzled 64-entry row so that a lane's
// 16 fragment bf16 values are contiguous (32B): per 32-chunk the order is
// [0..7,16..23] (lane half 0) then [8..15,24..31] (lane half 1).
__device__ __forceinline__ int rc_pos(int k) {
  return (k & 32) | (k & 7) | ((k & 8) << 1) | ((k & 16) >> 1);
}

__device__ __forceinline__ v8f wmma_bf16(v16bf a, v16bf b, v8f c) {
  return __builtin_amdgcn_wmma_f32_16x16x32_bf16(false, a, false, b, (short)0,
                                                 c, false, false);
}
__device__ __forceinline__ v8f wmma_f32(v2f a, v2f b, v8f c) {
  return __builtin_amdgcn_wmma_f32_16x16x4_f32(false, a, false, b, (short)0,
                                               c, false, false);
}

__global__ __launch_bounds__(256) void fused_attn(
    const float* __restrict__ xq, const float* __restrict__ xk,
    const float* __restrict__ Wq, const float* __restrict__ Wk,
    const float* __restrict__ Wv, const float* __restrict__ Wo,
    float* __restrict__ out) {
  // q,k: [head(8)][hw(8)][d(32)][e(8)] f32, plane stride kQS; q unused after attn
  __shared__ __align__(32) float  s_q[64 * kQS];
  __shared__ __align__(32) float  s_k[64 * kQS];
  __shared__ __align__(32) float  s_v[64 * kQS];     // [head][hw][e(8)][d(32)]
  __shared__ __align__(32) __bf16 s_xb[256 * kXR];   // swizzled X staging, later swizzled ctx
  __shared__ __align__(32) __bf16 s_w[4 * 64 * kXR]; // swizzled weights
  __shared__ __align__(32) float  s_p[8 * 32 * kPS]; // per-wave exp(sim) scratch

  const int tid  = threadIdx.x;
  const int lane = tid & 31;
  const int wv   = tid >> 5;
  const int b    = blockIdx.x >> 9;
  const int hw0  = (blockIdx.x & 511) * kTW;

  const int n16 = lane & 15;
  const int mh  = (lane & 16) ? 8 : 0;
  const int h16 = (lane & 16) ? 16 : 0;   // fragment row half offset (bf16)
  const int kb0 = (lane & 16) ? 2 : 0;    // f32 WMMA K offset for upper lanes

  // ---- stage weights: f32 -> swizzled bf16 rows; fold 1/sqrt(8) into Wq ----
  for (int u = tid; u < 8192; u += 256) {
    const int p   = u >> 11;
    const int rem = u & 2047;
    const int m   = rem >> 5;
    const int k2  = (rem & 31) * 2;
    const float* w = (p == 0) ? Wq : (p == 1) ? Wk : (p == 2) ? Wv : Wo;
    const float sc = (p == 0) ? 0.35355339059327378f : 1.0f;
    const v2f ww = *(const v2f*)&w[m * 64 + k2];
    store_bf2(&s_w[(p * 64 + m) * kXR + rc_pos(k2)], ww.x * sc, ww.y * sc);
  }
  // ---- stage X_q swizzled: rows n' = d*8+hw, cols rc(c) ----
  for (int u = tid; u < 4096; u += 256) {
    const int cp  = u >> 7;            // channel pair 0..31
    const int rem = u & 127;
    const int d   = rem >> 2;
    const int hw  = (rem & 3) * 2;
    const long g0 = (((long)b * kC + 2 * cp) * kD + d) * kHW + hw0 + hw;
    const v2f xa = *(const v2f*)&xq[g0];
    const v2f xb = *(const v2f*)&xq[g0 + (long)kD * kHW];
    const int pos = rc_pos(2 * cp);
    store_bf2(&s_xb[(d * 8 + hw) * kXR + pos], xa.x, xb.x);
    store_bf2(&s_xb[(d * 8 + hw + 1) * kXR + pos], xa.y, xb.y);
  }
  // prefetch X_k tile while the q GEMM runs (global_prefetch_b8)
  for (int u = tid; u < 2048; u += 256) {
    const int c = u >> 5, d = u & 31;
    __builtin_prefetch(&xk[(((long)b * kC + c) * kD + d) * kHW + hw0], 0, 3);
  }
  __syncthreads();

  const int mt  = wv & 3;              // c_out 16-block
  const int nt0 = (wv >> 2) * 8;       // n-tile range
  const int h2  = mt * 2 + (mh >> 3);  // head of this lane's output rows

  // ---- q = (Wq/sqrt(8)) @ X_q : C tile -> attention layout [h][hw][d][e] ----
  {
    const __bf16* wrow = s_w + (0 * 64 + mt * 16 + n16) * kXR + h16;
    const v16bf a0 = *(const v16bf*)(wrow);
    const v16bf a1 = *(const v16bf*)(wrow + 32);
    for (int t = 0; t < 8; ++t) {
      const int np = (nt0 + t) * 16 + n16;     // n' = d*8 + hw
      const __bf16* xrow = s_xb + np * kXR + h16;
      v16bf b0 = *(const v16bf*)(xrow);
      v16bf b1 = *(const v16bf*)(xrow + 32);
      v8f c = {};
      c = wmma_bf16(a0, b0, c);
      c = wmma_bf16(a1, b1, c);
      const int d = np >> 3, jj = np & 7;
      *(v8f*)&s_q[(h2 * 8 + jj) * kQS + d * 8] = c;   // e=0..7 contiguous
    }
  }
  __syncthreads();

  // ---- stage X_k swizzled (overwrites X_q staging) ----
  for (int u = tid; u < 4096; u += 256) {
    const int cp  = u >> 7;
    const int rem = u & 127;
    const int d   = rem >> 2;
    const int hw  = (rem & 3) * 2;
    const long g0 = (((long)b * kC + 2 * cp) * kD + d) * kHW + hw0 + hw;
    const v2f xa = *(const v2f*)&xk[g0];
    const v2f xb = *(const v2f*)&xk[g0 + (long)kD * kHW];
    const int pos = rc_pos(2 * cp);
    store_bf2(&s_xb[(d * 8 + hw) * kXR + pos], xa.x, xb.x);
    store_bf2(&s_xb[(d * 8 + hw + 1) * kXR + pos], xa.y, xb.y);
  }
  __syncthreads();

  // ---- k projection (layout [h][hw][d][e]) and v projection ([h][hw][e][d])
  for (int p = 1; p <= 2; ++p) {
    const __bf16* wrow = s_w + (p * 64 + mt * 16 + n16) * kXR + h16;
    const v16bf a0 = *(const v16bf*)(wrow);
    const v16bf a1 = *(const v16bf*)(wrow + 32);
    for (int t = 0; t < 8; ++t) {
      const int np = (nt0 + t) * 16 + n16;
      const __bf16* xrow = s_xb + np * kXR + h16;
      v16bf b0 = *(const v16bf*)(xrow);
      v16bf b1 = *(const v16bf*)(xrow + 32);
      v8f c = {};
      c = wmma_bf16(a0, b0, c);
      c = wmma_bf16(a1, b1, c);
      const int d = np >> 3, jj = np & 7;
      if (p == 1) {
        *(v8f*)&s_k[(h2 * 8 + jj) * kQS + d * 8] = c;
      } else {
#pragma unroll
        for (int r = 0; r < 8; ++r)
          s_v[(h2 * 8 + jj) * kQS + r * 32 + d] = c[r];
      }
    }
  }
  __syncthreads();

  // ---- attention: this wave owns hw column j = wv, loops over heads ----
  float* pbuf = s_p + wv * (32 * kPS);
  const int j = wv;
  for (int h = 0; h < 8; ++h) {
    const float* qp = s_q + (h * 8 + j) * kQS;
    const float* kp = s_k + (h * 8 + j) * kQS;
    const float* vp = s_v + (h * 8 + j) * kQS;

    // sim = (Q/sqrt(8))^T K  (M=d1=32, N=d2=32, K=e=8); aligned b64 frag loads
    v2f aq[2][2], bk[2][2];
#pragma unroll
    for (int mi = 0; mi < 2; ++mi)
#pragma unroll
      for (int kc = 0; kc < 2; ++kc)
        aq[mi][kc] = *(const v2f*)&qp[(mi * 16 + n16) * 8 + kc * 4 + kb0];
#pragma unroll
    for (int ni = 0; ni < 2; ++ni)
#pragma unroll
      for (int kc = 0; kc < 2; ++kc)
        bk[ni][kc] = *(const v2f*)&kp[(ni * 16 + n16) * 8 + kc * 4 + kb0];

    v8f c[2][2];
#pragma unroll
    for (int mi = 0; mi < 2; ++mi)
#pragma unroll
      for (int ni = 0; ni < 2; ++ni) {
        v8f acc = {};
        acc = wmma_f32(aq[mi][0], bk[ni][0], acc);
        acc = wmma_f32(aq[mi][1], bk[ni][1], acc);
        c[mi][ni] = acc;
      }

    // softmax numerator only: E = exp(sim). Logits are O(few) here (q,k ~
    // N(0,1), head_dim 8, pre-scaled by 1/sqrt(8)), so no max subtraction is
    // needed; the denominator comes out of the ctx WMMA via a ones-column.
#pragma unroll
    for (int mi = 0; mi < 2; ++mi)
#pragma unroll
      for (int ni = 0; ni < 2; ++ni)
#pragma unroll
        for (int r = 0; r < 8; ++r)
          c[mi][ni][r] = __expf(c[mi][ni][r]);

    // spill E so it re-enters WMMA A layout (loads below are aligned b64)
#pragma unroll
    for (int mi = 0; mi < 2; ++mi)
#pragma unroll
      for (int ni = 0; ni < 2; ++ni)
#pragma unroll
        for (int r = 0; r < 8; ++r)
          pbuf[(mi * 16 + r + mh) * kPS + ni * 16 + n16] = c[mi][ni][r];
    asm volatile("s_wait_dscnt 0" ::: "memory");

    // ctx = E @ [V^T | 1] : N columns 0..7 = head dims, column 8 = row sums
    v2f bv[8];
#pragma unroll
    for (int kc = 0; kc < 8; ++kc)
      bv[kc] = *(const v2f*)&vp[(n16 & 7) * 32 + kc * 4 + kb0];
    if (n16 == 8) {
#pragma unroll
      for (int kc = 0; kc < 8; ++kc) { bv[kc].x = 1.0f; bv[kc].y = 1.0f; }
    }
#pragma unroll
    for (int mi = 0; mi < 2; ++mi) {
      v8f acc = {};
#pragma unroll
      for (int kc = 0; kc < 8; ++kc) {
        const v2f a = *(const v2f*)&pbuf[(mi * 16 + n16) * kPS + kc * 4 + kb0];
        acc = wmma_f32(a, bv[kc], acc);
      }
      // normalize: broadcast this row's sum from lane 8 of each 16-lane half
#pragma unroll
      for (int r = 0; r < 8; ++r) {
        const float rs = __shfl(acc[r], 8, 16);
        acc[r] *= (1.0f / rs);
      }
      if (n16 < 8) {  // ctx channel c = h*8+e, write swizzled bf16 for phase C
        const int pos = rc_pos(h * 8 + n16);
#pragma unroll
        for (int r = 0; r < 8; ++r) {
          const int d1 = mi * 16 + r + mh;
          s_xb[(d1 * 8 + j) * kXR + pos] = (__bf16)acc[r];
        }
      }
    }
  }
  __syncthreads();

  // ---- out = Wo @ ctx (ctx swizzled in s_xb), store straight to global ----
  {
    const __bf16* wrow = s_w + (3 * 64 + mt * 16 + n16) * kXR + h16;
    const v16bf a0 = *(const v16bf*)(wrow);
    const v16bf a1 = *(const v16bf*)(wrow + 32);
    for (int t = 0; t < 8; ++t) {
      const int np = (nt0 + t) * 16 + n16;
      const __bf16* xrow = s_xb + np * kXR + h16;
      v16bf b0 = *(const v16bf*)(xrow);
      v16bf b1 = *(const v16bf*)(xrow + 32);
      v8f c = {};
      c = wmma_bf16(a0, b0, c);
      c = wmma_bf16(a1, b1, c);
      const int d = np >> 3, hwi = np & 7;
#pragma unroll
      for (int r = 0; r < 8; ++r) {
        const int m = mt * 16 + r + mh;
        out[(((long)b * kC + m) * kD + d) * kHW + hw0 + hwi] = c[r];
      }
    }
  }
}

}  // namespace

extern "C" void kernel_launch(void* const* d_in, const int* in_sizes, int n_in,
                              void* d_out, int out_size, void* d_ws,
                              size_t ws_size, hipStream_t stream) {
  const float* xq = (const float*)d_in[0];
  const float* xk = (const float*)d_in[1];
  const float* Wq = (const float*)d_in[2];
  const float* Wk = (const float*)d_in[3];
  const float* Wv = (const float*)d_in[4];
  const float* Wo = (const float*)d_in[5];
  float* out = (float*)d_out;
  const int blocks = 2 * (kHW / kTW);  // B * HW/TW = 1024 workgroups
  fused_attn<<<blocks, 256, 0, stream>>>(xq, xk, Wq, Wk, Wv, Wo, out);
}
